// RFS_18588618457656
// MI455X (gfx1250) — compile-verified, matrix operand order checked by reading, not run on hardware
//
#include <hip/hip_runtime.h>

// ---------------------------------------------------------------------------
// Relation-Network forward for MI455X (gfx1250, wave32, WMMA).
//   Stage 1: 4x (direct conv fp32 -> ReLU -> batch-stats reduce -> normalize)
//   Stage 2: build keys[B,26,20]/vals[B,26,16] (+coords)
//   Stage 3: pack GRU weights into f16 WMMA-B fragment layout (once per call)
//   Stage 4: fused persistent scan kernel: 12 steps of (GRU,GRU,attn) +
//            12 steps of (GRU,GRU) + log_softmax head.  16 batch rows / WG,
//            all GEMMs via v_wmma_f32_16x16x32_f16, state lives in LDS.
//            Wave id is readfirstlane'd so tile dispatch is scalar-branched
//            (EXEC provably all-ones at every WMMA), and packed weights are
//            prefetched (global_prefetch_b8) before the sequential recurrence.
// ---------------------------------------------------------------------------

typedef __attribute__((ext_vector_type(16))) _Float16 v16h;
typedef __attribute__((ext_vector_type(8)))  _Float16 v8h;
typedef __attribute__((ext_vector_type(8)))  float    v8f;

#define NB    1024
#define HD    128
#define SEQN  12

// A/B fragment K-mapping for 16-bit WMMA (16x16x32), per ISA 7.12.2:
// lane L holds row/col (L&15); element j of the v16h maps to
// k = ((j>>3)<<4) + ((L&16)>>1) + (j&7).
__device__ __forceinline__ int kmap(int j, int lane) {
  return ((j >> 3) << 4) + ((lane & 16) >> 1) + (j & 7);
}

__device__ __forceinline__ v8f wmma16(v16h a, v16h b, v8f c) {
  return __builtin_amdgcn_wmma_f32_16x16x32_f16(false, a, false, b, (short)0, c,
                                                false, false);
}

__device__ __forceinline__ float sigm(float x) { return 1.f / (1.f + __expf(-x)); }

// ------------------------------- conv stack --------------------------------

__global__ void rn_conv_relu(const float* __restrict__ x, const float* __restrict__ w,
                             const float* __restrict__ bias, float* __restrict__ y,
                             int Cin, int Hin, int Win, int Hout, int Wout) {
  long idx = (long)blockIdx.x * blockDim.x + threadIdx.x;
  long total = (long)NB * 24 * Hout * Wout;
  if (idx >= total) return;
  int wo = (int)(idx % Wout); long t = idx / Wout;
  int ho = (int)(t % Hout);  t /= Hout;
  int co = (int)(t % 24);    int bb = (int)(t / 24);
  float acc = bias[co];
  int hi0 = ho * 2 - 1, wi0 = wo * 2 - 1;
  for (int ci = 0; ci < Cin; ci++) {
    const float* xp = x + ((long)(bb * Cin + ci) * Hin) * Win;
    const float* wp = w + ((co * Cin + ci) * 9);
    for (int kh = 0; kh < 3; kh++) {
      int hi = hi0 + kh; if ((unsigned)hi >= (unsigned)Hin) continue;
      for (int kw = 0; kw < 3; kw++) {
        int wi = wi0 + kw; if ((unsigned)wi >= (unsigned)Win) continue;
        acc += xp[(long)hi * Win + wi] * wp[kh * 3 + kw];
      }
    }
  }
  y[idx] = fmaxf(acc, 0.f);
}

// one block per (b, c) plane; tree reduce; atomic into 24x2 stats
__global__ void rn_bn_stats(const float* __restrict__ y, float* __restrict__ stats,
                            int HW) {
  int plane = blockIdx.x;            // b*24 + c
  int c = plane % 24;
  const float* p = y + (long)plane * HW;
  float s = 0.f, q = 0.f;
  for (int i = threadIdx.x; i < HW; i += blockDim.x) {
    float v = p[i]; s += v; q += v * v;
  }
  __shared__ float rs[256], rq[256];
  rs[threadIdx.x] = s; rq[threadIdx.x] = q;
  __syncthreads();
  for (int off = 128; off > 0; off >>= 1) {
    if ((int)threadIdx.x < off) {
      rs[threadIdx.x] += rs[threadIdx.x + off];
      rq[threadIdx.x] += rq[threadIdx.x + off];
    }
    __syncthreads();
  }
  if (threadIdx.x == 0) {
    atomicAdd(&stats[c * 2 + 0], rs[0]);
    atomicAdd(&stats[c * 2 + 1], rq[0]);
  }
}

__global__ void rn_bn_apply(float* __restrict__ y, const float* __restrict__ stats,
                            const float* __restrict__ g, const float* __restrict__ bt,
                            long total, int HW, float invcnt) {
  long i = (long)blockIdx.x * blockDim.x + threadIdx.x;
  if (i >= total) return;
  int c = (int)((i / HW) % 24);
  float mu  = stats[c * 2 + 0] * invcnt;
  float var = stats[c * 2 + 1] * invcnt - mu * mu;
  y[i] = (y[i] - mu) * rsqrtf(var + 1e-5f) * g[c] + bt[c];
}

// ------------------------------ keys / values ------------------------------

__global__ void rn_build_kv(const float* __restrict__ x3, const float* __restrict__ kb,
                            float* __restrict__ keys, float* __restrict__ vals) {
  int idx = blockIdx.x * blockDim.x + threadIdx.x;
  if (idx >= NB * 26) return;
  int e = idx % 26, b = idx / 26;
  float* kr = keys + (long)idx * 20;
  float* vr = vals + (long)idx * 16;
  if (e == 25) {
    for (int k = 0; k < 20; k++) kr[k] = kb[k];
    for (int j = 0; j < 16; j++) vr[j] = 0.f;
  } else {
    int i5 = e / 5, j5 = e % 5;
    float coord[6] = {(i5 - 2) * 0.5f, (float)(i5 & 1), i5 > 0 ? 1.f : 0.f,
                      (j5 - 2) * 0.5f, (float)(j5 & 1), j5 > 0 ? 1.f : 0.f};
    for (int k = 0; k < 14; k++) kr[k] = x3[((long)b * 24 + k) * 25 + e];
    for (int k = 0; k < 6;  k++) kr[14 + k] = coord[k];
    for (int j = 0; j < 10; j++) vr[j] = x3[((long)b * 24 + 14 + j) * 25 + e];
    for (int j = 0; j < 6;  j++) vr[10 + j] = coord[j];
  }
}

// ------------------------------ weight packing -----------------------------
// W is [N=384, K] row-major; GEMM computes X @ W^T, so B[k][n] = W[n][k].
// Pack B into per-(tile,chunk,lane) v16h fragments; K padded to 32*KC with 0.

__global__ void rn_pack_w(const float* __restrict__ W, _Float16* __restrict__ P,
                          int Kreal, int KC) {
  int idx = blockIdx.x * blockDim.x + threadIdx.x;
  int total = 24 * KC * 512;             // NT=24 tiles, 32 lanes * 16 elems
  if (idx >= total) return;
  int j    = idx & 15;
  int lane = (idx >> 4) & 31;
  int c    = (idx >> 9) % KC;
  int t    = idx / (512 * KC);
  int n = t * 16 + (lane & 15);
  int k = c * 32 + kmap(j, lane);
  float v = (k < Kreal) ? W[n * Kreal + k] : 0.f;
  P[idx] = (_Float16)v;
}

// packed offsets (in halves)
#define PK_R1_WIH 0
#define PK_R1_WHH 12288
#define PK_R2_WIH 61440
#define PK_R2_WHH 110592
#define PK_SQ_WIH 159744
#define PK_SQ_WHH 172032
#define PK_SA_WIH 221184
#define PK_SA_WHH 270336
#define PK_TOTAL  319488

// -------------------------------- scan kernel ------------------------------

struct ScanArgs {
  const float* qst;
  const float* keys;
  const float* vals;
  const _Float16* pk;
  const float *r1_bih, *r1_bhh, *r2_bih, *r2_bhh;
  const float *sq_bih, *sq_bhh, *sa_bih, *sa_bhh;
  const float *q_w, *q_b, *out_w, *out_b;
  const float *rnn1_pad, *rnn1_start, *rnn2_h0, *sq_pad, *sa_h0;
  float* out;
};

__device__ __forceinline__ v16h load_frag(const _Float16* base, int lane) {
  return *(const v16h*)(base + lane * 16);       // 32B aligned -> ds_load_b128 x2
}
__device__ __forceinline__ v16h load_halffrag(const _Float16* base, int lane) {
  v8h lo = *(const v8h*)(base + lane * 8);
  v16h r;
#pragma unroll
  for (int j = 0; j < 8; j++) r[j] = lo[j];
#pragma unroll
  for (int j = 0; j < 8; j++) r[8 + j] = (_Float16)0.f;
  return r;
}
__device__ __forceinline__ void store_acc(float* dst, int col0, v8f acc, int lane) {
  int m0 = (lane >> 4) << 3;
  int col = col0 + (lane & 15);
#pragma unroll
  for (int r = 0; r < 8; r++) dst[(m0 + r) * HD + col] = acc[r];
}

// One GRU cell's two GEMMs.  8 waves x 3 N-tiles cover N=384.
// r/z ranges accumulate gi+gh in one chain; n range stores gi,gh separately.
// `wave` is an SGPR (readfirstlane'd) so all tile branches are scalar and
// EXEC is all-ones at every WMMA.
__device__ __forceinline__ void gru_gemm(const _Float16* wih, const _Float16* whh,
                                         const _Float16* xf, int x_kc, bool x_half,
                                         const _Float16* hf,
                                         float* g_r, float* g_z, float* g_in,
                                         float* g_hn, int wave, int lane) {
#pragma unroll
  for (int tt = 0; tt < 3; tt++) {
    int T = wave * 3 + tt;                       // scalar: s_cmp/s_cbranch below
    v8f acc = {};
#pragma unroll
    for (int c = 0; c < 4; c++) {
      if (c >= x_kc) break;
      v16h a = x_half ? load_halffrag(xf, lane) : load_frag(xf + c * 512, lane);
      v16h b = *(const v16h*)(wih + ((T * x_kc + c) * 32 + lane) * 16);
      acc = wmma16(a, b, acc);
    }
    if (T < 16) {
#pragma unroll
      for (int c = 0; c < 4; c++) {
        v16h a = load_frag(hf + c * 512, lane);
        v16h b = *(const v16h*)(whh + ((T * 4 + c) * 32 + lane) * 16);
        acc = wmma16(a, b, acc);
      }
      store_acc((T < 8) ? g_r : g_z, (T & 7) * 16, acc, lane);
    } else {
      store_acc(g_in, (T - 16) * 16, acc, lane);
      v8f acc2 = {};
#pragma unroll
      for (int c = 0; c < 4; c++) {
        v16h a = load_frag(hf + c * 512, lane);
        v16h b = *(const v16h*)(whh + ((T * 4 + c) * 32 + lane) * 16);
        acc2 = wmma16(a, b, acc2);
      }
      store_acc(g_hn, (T - 16) * 16, acc2, lane);
    }
  }
}

// GRU gate math + writeback of new h (f32 row-major and f16 A-fragment layout)
__device__ __forceinline__ void gru_combine(const float* bih, const float* bhh,
                                            float* h, _Float16* hf,
                                            const float* g_r, const float* g_z,
                                            const float* g_in, const float* g_hn,
                                            int tid) {
#pragma unroll
  for (int p = 0; p < 8; p++) {
    int idx = tid + p * 256;
    int m = idx >> 7, k = idx & 127;
    float r = sigm(g_r[idx] + bih[k] + bhh[k]);
    float z = sigm(g_z[idx] + bih[128 + k] + bhh[128 + k]);
    float n = tanhf(g_in[idx] + bih[256 + k] + r * (g_hn[idx] + bhh[256 + k]));
    float hn = (1.f - z) * n + z * h[idx];
    h[idx] = hn;
    int c = k >> 5, kk = k & 31;
    int lane = ((kk & 8) ? 16 : 0) + m;
    int j = ((kk >> 4) << 3) | (kk & 7);
    hf[(c * 32 + lane) * 16 + j] = (_Float16)hn;
  }
}

__global__ void __launch_bounds__(256) rn_scan(ScanArgs args) {
  __shared__ float g_r[16 * HD];
  __shared__ float g_z[16 * HD];
  __shared__ float g_in[16 * HD];
  __shared__ float g_hn[16 * HD];
  __shared__ float h1[16 * HD];
  __shared__ float h2[16 * HD];
  __shared__ __align__(32) _Float16 h1f[4 * 32 * 16];
  __shared__ __align__(32) _Float16 h2f[4 * 32 * 16];
  __shared__ __align__(16) _Float16 vf[32 * 8];          // current v (K=16, low half)
  __shared__ __align__(16) _Float16 valf[SEQN * 32 * 8]; // per-step values
  float* qbuf = g_r;                                     // reuse scratch for q[16][20]

  int tid = threadIdx.x, lane = tid & 31;
  int wave = __builtin_amdgcn_readfirstlane(tid >> 5);   // SGPR wave id
  int b0 = blockIdx.x * 16;

  // Warm WGP$/L2 with the packed weights this WG will re-read 24x in the
  // sequential recurrence (lowers to global_prefetch_b8; no counter traffic).
  for (size_t off = (size_t)tid * 256; off < (size_t)PK_TOTAL * 2; off += 256u * 256u)
    __builtin_prefetch((const char*)args.pk + off, 0, 3);

  // ---- init scan-1 state: h1 = [qst | rnn1_pad], h2 = rnn2_h0, v = rnn1_start
#pragma unroll
  for (int p = 0; p < 8; p++) {
    int idx = tid + p * 256;
    int m = idx >> 7, k = idx & 127;
    float a = (k < 11) ? args.qst[(b0 + m) * 11 + k] : args.rnn1_pad[k - 11];
    float b = args.rnn2_h0[k];
    h1[idx] = a; h2[idx] = b;
    int c = k >> 5, kk = k & 31;
    int ln = ((kk & 8) ? 16 : 0) + m;
    int j = ((kk >> 4) << 3) | (kk & 7);
    h1f[(c * 32 + ln) * 16 + j] = (_Float16)a;
    h2f[(c * 32 + ln) * 16 + j] = (_Float16)b;
  }
  {
    int m = tid >> 4, kk = tid & 15;
    int ln = ((kk & 8) ? 16 : 0) + m;
    vf[ln * 8 + (kk & 7)] = (_Float16)args.rnn1_start[kk];
  }
  __syncthreads();

  // ---------------------------- scan 1: 12 steps ----------------------------
#pragma unroll 1
  for (int s = 0; s < SEQN; s++) {
    gru_gemm(args.pk + PK_R1_WIH, args.pk + PK_R1_WHH, vf, 1, true, h1f,
             g_r, g_z, g_in, g_hn, wave, lane);
    __syncthreads();
    gru_combine(args.r1_bih, args.r1_bhh, h1, h1f, g_r, g_z, g_in, g_hn, tid);
    __syncthreads();
    gru_gemm(args.pk + PK_R2_WIH, args.pk + PK_R2_WHH, h1f, 4, false, h2f,
             g_r, g_z, g_in, g_hn, wave, lane);
    __syncthreads();
    gru_combine(args.r2_bih, args.r2_bhh, h2, h2f, g_r, g_z, g_in, g_hn, tid);
    __syncthreads();
    // q = h2 @ q_w^T + q_b   (16x20, VALU tail)
    for (int idx = tid; idx < 320; idx += 256) {
      int m = idx / 20, o = idx % 20;
      float a = args.q_b[o];
      for (int k = 0; k < HD; k++) a += h2[m * HD + k] * args.q_w[o * HD + k];
      qbuf[m * 20 + o] = a;
    }
    __syncthreads();
    // attention: softmax over 26 keys, mix vals -> new v
    if (tid < 16) {
      int m = tid, b = b0 + m;
      float lg[26]; float mx = -1e30f;
      for (int e = 0; e < 26; e++) {
        const float* kr = args.keys + ((long)b * 26 + e) * 20;
        float a = 0.f;
        for (int k = 0; k < 20; k++) a += kr[k] * qbuf[m * 20 + k];
        lg[e] = a; mx = fmaxf(mx, a);
      }
      float sum = 0.f;
      for (int e = 0; e < 26; e++) { lg[e] = __expf(lg[e] - mx); sum += lg[e]; }
      float inv = 1.f / sum;
      float vn[16];
      for (int j = 0; j < 16; j++) vn[j] = 0.f;
      for (int e = 0; e < 26; e++) {
        const float* vr = args.vals + ((long)b * 26 + e) * 16;
        float w = lg[e] * inv;
        for (int j = 0; j < 16; j++) vn[j] += w * vr[j];
      }
      for (int kk = 0; kk < 16; kk++) {
        int ln = ((kk & 8) ? 16 : 0) + m;
        _Float16 hv = (_Float16)vn[kk];
        vf[ln * 8 + (kk & 7)] = hv;
        valf[(s * 32 + ln) * 8 + (kk & 7)] = hv;
      }
    }
    __syncthreads();
  }

  // ---- init scan-2 state: h1 = [qst | sq_pad], h2 = sa_h0
#pragma unroll
  for (int p = 0; p < 8; p++) {
    int idx = tid + p * 256;
    int m = idx >> 7, k = idx & 127;
    float a = (k < 11) ? args.qst[(b0 + m) * 11 + k] : args.sq_pad[k - 11];
    float b = args.sa_h0[k];
    h1[idx] = a; h2[idx] = b;
    int c = k >> 5, kk = k & 31;
    int ln = ((kk & 8) ? 16 : 0) + m;
    int j = ((kk >> 4) << 3) | (kk & 7);
    h1f[(c * 32 + ln) * 16 + j] = (_Float16)a;
    h2f[(c * 32 + ln) * 16 + j] = (_Float16)b;
  }
  __syncthreads();

  // ---------------------------- scan 2: 12 steps ----------------------------
#pragma unroll 1
  for (int s = 0; s < SEQN; s++) {
    gru_gemm(args.pk + PK_SQ_WIH, args.pk + PK_SQ_WHH, valf + s * 256, 1, true,
             h1f, g_r, g_z, g_in, g_hn, wave, lane);
    __syncthreads();
    gru_combine(args.sq_bih, args.sq_bhh, h1, h1f, g_r, g_z, g_in, g_hn, tid);
    __syncthreads();
    gru_gemm(args.pk + PK_SA_WIH, args.pk + PK_SA_WHH, h1f, 4, false, h2f,
             g_r, g_z, g_in, g_hn, wave, lane);
    __syncthreads();
    gru_combine(args.sa_bih, args.sa_bhh, h2, h2f, g_r, g_z, g_in, g_hn, tid);
    __syncthreads();
  }

  // ---- head: log_softmax(ha @ out_w^T + out_b)
  if (tid < 16) {
    int m = tid, b = b0 + m;
    float lg[10]; float mx = -1e30f;
    for (int o = 0; o < 10; o++) {
      float a = args.out_b[o];
      for (int k = 0; k < HD; k++) a += h2[m * HD + k] * args.out_w[o * HD + k];
      lg[o] = a; mx = fmaxf(mx, a);
    }
    float sum = 0.f;
    for (int o = 0; o < 10; o++) sum += __expf(lg[o] - mx);
    float lse = mx + __logf(sum);
    for (int o = 0; o < 10; o++) args.out[b * 10 + o] = lg[o] - lse;
  }
}

// --------------------------------- launcher --------------------------------

extern "C" void kernel_launch(void* const* d_in, const int* in_sizes, int n_in,
                              void* d_out, int out_size, void* d_ws, size_t ws_size,
                              hipStream_t stream) {
  (void)in_sizes; (void)n_in; (void)out_size; (void)ws_size;
  const float* img = (const float*)d_in[0];
  const float* qst = (const float*)d_in[1];
  const float* convw[4] = {(const float*)d_in[2], (const float*)d_in[6],
                           (const float*)d_in[10], (const float*)d_in[14]};
  const float* convb[4] = {(const float*)d_in[3], (const float*)d_in[7],
                           (const float*)d_in[11], (const float*)d_in[15]};
  const float* bng[4]   = {(const float*)d_in[4], (const float*)d_in[8],
                           (const float*)d_in[12], (const float*)d_in[16]};
  const float* bnb[4]   = {(const float*)d_in[5], (const float*)d_in[9],
                           (const float*)d_in[13], (const float*)d_in[17]};
  const float* k_blank    = (const float*)d_in[18];
  const float* rnn1_pad   = (const float*)d_in[19];
  const float* rnn1_start = (const float*)d_in[20];
  const float* rnn2_h0    = (const float*)d_in[21];
  const float* r1_wih = (const float*)d_in[22]; const float* r1_whh = (const float*)d_in[23];
  const float* r1_bih = (const float*)d_in[24]; const float* r1_bhh = (const float*)d_in[25];
  const float* r2_wih = (const float*)d_in[26]; const float* r2_whh = (const float*)d_in[27];
  const float* r2_bih = (const float*)d_in[28]; const float* r2_bhh = (const float*)d_in[29];
  const float* sq_wih = (const float*)d_in[30]; const float* sq_whh = (const float*)d_in[31];
  const float* sq_bih = (const float*)d_in[32]; const float* sq_bhh = (const float*)d_in[33];
  const float* sa_wih = (const float*)d_in[34]; const float* sa_whh = (const float*)d_in[35];
  const float* sa_bih = (const float*)d_in[36]; const float* sa_bhh = (const float*)d_in[37];
  const float* q_w   = (const float*)d_in[38]; const float* q_b   = (const float*)d_in[39];
  const float* sq_pad = (const float*)d_in[40]; const float* sa_h0 = (const float*)d_in[41];
  const float* out_w = (const float*)d_in[42]; const float* out_b = (const float*)d_in[43];

  char* wsb = (char*)d_ws;
  size_t OFF_BUF0 = 0;
  size_t OFF_BUF1 = OFF_BUF0 + 141950976ull;  // 1024*24*38*38*4
  size_t OFF_BUF2 = OFF_BUF1 + 35487744ull;   // 1024*24*19*19*4
  size_t OFF_BUF3 = OFF_BUF2 + 9830400ull;    // 1024*24*10*10*4
  size_t OFF_KEYS = OFF_BUF3 + 2457600ull;    // 1024*24*5*5*4
  size_t OFF_VALS = OFF_KEYS + 2129920ull;    // 1024*26*20*4
  size_t OFF_STAT = OFF_VALS + 1703936ull;    // 1024*26*16*4
  size_t OFF_PACK = OFF_STAT + 768ull;        // 4*24*2*4 stats

  float* bufs[4] = {(float*)(wsb + OFF_BUF0), (float*)(wsb + OFF_BUF1),
                    (float*)(wsb + OFF_BUF2), (float*)(wsb + OFF_BUF3)};
  float* keys = (float*)(wsb + OFF_KEYS);
  float* vals = (float*)(wsb + OFF_VALS);
  float* stats = (float*)(wsb + OFF_STAT);
  _Float16* pk = (_Float16*)(wsb + OFF_PACK);

  hipMemsetAsync(stats, 0, 768, stream);

  const int Cin[4]  = {3, 24, 24, 24};
  const int Hin[4]  = {75, 38, 19, 10};
  const int Hout[4] = {38, 19, 10, 5};
  const float* src = img;
  for (int l = 0; l < 4; l++) {
    int HW = Hout[l] * Hout[l];
    long total = (long)NB * 24 * HW;
    int blocks = (int)((total + 255) / 256);
    rn_conv_relu<<<blocks, 256, 0, stream>>>(src, convw[l], convb[l], bufs[l],
                                             Cin[l], Hin[l], Hin[l], Hout[l], Hout[l]);
    rn_bn_stats<<<NB * 24, 256, 0, stream>>>(bufs[l], stats + l * 48, HW);
    rn_bn_apply<<<blocks, 256, 0, stream>>>(bufs[l], stats + l * 48, bng[l], bnb[l],
                                            total, HW, 1.f / (float)((long)NB * HW));
    src = bufs[l];
  }

  rn_build_kv<<<(NB * 26 + 255) / 256, 256, 0, stream>>>(bufs[3], k_blank, keys, vals);

  // pack GRU weights into WMMA-B fragment order (f16)
  rn_pack_w<<<48, 256, 0, stream>>>(r1_wih, pk + PK_R1_WIH, 16, 1);
  rn_pack_w<<<192, 256, 0, stream>>>(r1_whh, pk + PK_R1_WHH, 128, 4);
  rn_pack_w<<<192, 256, 0, stream>>>(r2_wih, pk + PK_R2_WIH, 128, 4);
  rn_pack_w<<<192, 256, 0, stream>>>(r2_whh, pk + PK_R2_WHH, 128, 4);
  rn_pack_w<<<48, 256, 0, stream>>>(sq_wih, pk + PK_SQ_WIH, 16, 1);
  rn_pack_w<<<192, 256, 0, stream>>>(sq_whh, pk + PK_SQ_WHH, 128, 4);
  rn_pack_w<<<192, 256, 0, stream>>>(sa_wih, pk + PK_SA_WIH, 128, 4);
  rn_pack_w<<<192, 256, 0, stream>>>(sa_whh, pk + PK_SA_WHH, 128, 4);

  ScanArgs a;
  a.qst = qst; a.keys = keys; a.vals = vals; a.pk = pk;
  a.r1_bih = r1_bih; a.r1_bhh = r1_bhh; a.r2_bih = r2_bih; a.r2_bhh = r2_bhh;
  a.sq_bih = sq_bih; a.sq_bhh = sq_bhh; a.sa_bih = sa_bih; a.sa_bhh = sa_bhh;
  a.q_w = q_w; a.q_b = q_b; a.out_w = out_w; a.out_b = out_b;
  a.rnn1_pad = rnn1_pad; a.rnn1_start = rnn1_start; a.rnn2_h0 = rnn2_h0;
  a.sq_pad = sq_pad; a.sa_h0 = sa_h0;
  a.out = (float*)d_out;
  rn_scan<<<NB / 16, 256, 0, stream>>>(a);
}
